// SingleConv2d_5428838662845
// MI455X (gfx1250) — matrix operand order repaired and checked by reading, not compiled, not run on hardware
//
#include <hip/hip_runtime.h>

// ---------------------------------------------------------------------------
// Binary-weight conv2d (Hubara binarize -> conv3x3 same) as halo-tiled
// implicit GEMM on CDNA5 V_WMMA_F32_16X16X32_BF16.
//   Block: 2 output rows x 56 cols (112 spatial, GEMM N) x 256 couts (GEMM M)
//   K loop: 8 cin-blocks of 32; 9 taps replayed from one LDS halo tile.
// ---------------------------------------------------------------------------

typedef __attribute__((ext_vector_type(16))) __bf16 v16bf;
typedef __attribute__((ext_vector_type(8)))  float  v8f;

#define NIMG   32
#define C_IN   256
#define C_OUT  256
#define HH     56
#define WW     56
#define HW     (HH * WW)            // 3136
#define KTAPS  9
#define KBLK   32                   // K per WMMA
#define NKB    (C_IN / KBLK)        // 8 cin blocks

#define TROWS  2                    // output rows per block
#define HROWS  4                    // halo rows staged (y-1 .. y+2)
#define HCOLS  64                   // halo cols staged (x = -1 .. 62, zero-padded)
#define BLK_S  (TROWS * WW)         // 112 spatial positions (7 n-tiles)
#define NTILES 7
#define COLSTR 80                   // LDS bytes per halo column: 64B (32 bf16) + 16B pad
#define LDSBUF (HROWS * HCOLS * COLSTR)   // 20480 B per buffer

// Packed binarized weights, A-fragment-ready (bf16 bits in ushort):
//   pk[t][kb][cb(16)][lane(32)][e(16)]
// lane<16 : row M=lane,    e->K = (e<8)? e    : e+8    (K 0..7,16..23)
// lane>=16: row M=lane-16, e->K = (e<8)? e+8  : e+16   (K 8..15,24..31)
__global__ void binarize_pack_kernel(const float* __restrict__ Wsrc,
                                     unsigned short* __restrict__ pk) {
    int idx = blockIdx.x * 256 + threadIdx.x;
    int e    = idx & 15;
    int lane = (idx >> 4) & 31;
    int cb   = (idx >> 9) & 15;
    int kb   = (idx >> 13) & 7;
    int t    = idx >> 16;                       // 0..8
    if (t >= KTAPS) return;
    int m    = lane & 15;
    int half = lane >> 4;
    int K    = half ? ((e < 8) ? e + 8 : e + 16)
                    : ((e < 8) ? e     : e + 8);
    int co = cb * 16 + m;
    int ci = kb * KBLK + K;
    float w = Wsrc[(co * C_IN + ci) * KTAPS + t];
    pk[idx] = (w >= 0.0f) ? (unsigned short)0x3F80u : (unsigned short)0xBF80u;
}

__device__ __forceinline__ unsigned int pack_bf16_pair(float lo, float hi) {
    unsigned int ulo = __float_as_uint(lo);
    unsigned int uhi = __float_as_uint(hi);
    ulo = (ulo + 0x7FFFu + ((ulo >> 16) & 1u)) >> 16;   // RNE f32->bf16
    uhi = (uhi + 0x7FFFu + ((uhi >> 16) & 1u)) >> 16;
    return ulo | (uhi << 16);
}

__launch_bounds__(256, 1)
__global__ void bconv_wmma_kernel(const float* __restrict__ xin,
                                  const unsigned short* __restrict__ pk,
                                  float* __restrict__ out) {
    __shared__ char lds[2 * LDSBUF];             // 40 KB double-buffered halo tile

    const int tid  = threadIdx.x;
    const int lane = tid & 31;
    const int wave = tid >> 5;                   // 8 waves, each owns 32 couts

    const int bx   = blockIdx.x;                 // 32 imgs * 28 row-tiles
    const int nimg = bx / (HH / TROWS);
    const int y0   = (bx - nimg * (HH / TROWS)) * TROWS;

    // ---- fill coordinates: one halo column per thread ----
    const int frow = tid >> 6;                   // 0..3  -> y = y0-1 .. y0+2
    const int fx   = (tid & 63) - 1;             // -1 .. 62
    const int ys   = y0 + frow - 1;
    const bool vb  = ((unsigned)ys < (unsigned)HH) && ((unsigned)fx < (unsigned)WW);
    const float* fsrc = xin + (size_t)nimg * C_IN * HW + (vb ? ys * WW + fx : 0);

    // ---- per-lane B-fragment column bases (constant over kb/taps) ----
    int bcol[NTILES];
    #pragma unroll
    for (int ni = 0; ni < NTILES; ni++) {
        int c  = ni * 16 + (lane & 15);          // 0..111
        int yl = (c >= WW) ? 1 : 0;
        int xl = c - WW * yl;
        bcol[ni] = (yl * HCOLS + xl) * COLSTR + (lane >> 4) * 16;
    }

    v8f acc[2][NTILES] = {};                     // 32 couts x 112 spatial
    float stg[32];                               // global->LDS staging (one cin block)

    auto load_stage = [&](int kb) {
        const float* src = fsrc + (size_t)(kb * KBLK) * HW;
        #pragma unroll
        for (int j = 0; j < 32; j++)
            stg[j] = vb ? src[(size_t)j * HW] : 0.0f;
    };

    auto store_stage = [&](int buf) {
        char* base = lds + buf * LDSBUF + tid * COLSTR;   // this thread's column
        #pragma unroll
        for (int q = 0; q < 4; q++) {
            uint4 v;
            v.x = pack_bf16_pair(stg[q * 8 + 0], stg[q * 8 + 1]);
            v.y = pack_bf16_pair(stg[q * 8 + 2], stg[q * 8 + 3]);
            v.z = pack_bf16_pair(stg[q * 8 + 4], stg[q * 8 + 5]);
            v.w = pack_bf16_pair(stg[q * 8 + 6], stg[q * 8 + 7]);
            *(uint4*)(base + q * 16) = v;
        }
    };

    auto compute = [&](int kb, int buf) {
        const char* lbase = lds + buf * LDSBUF;
        #pragma unroll
        for (int t = 0; t < KTAPS; t++) {
            const int kh = t / 3, kw = t - kh * 3;
            const int tapoff = (kh * HCOLS + kw) * COLSTR;   // tap = LDS offset
            // A fragments: packed binarized weights (L2-hot, 1.2 MB total)
            v16bf af[2];
            #pragma unroll
            for (int mi = 0; mi < 2; mi++) {
                int cbi = wave * 2 + mi;                      // 16-cout block
                const unsigned short* ap =
                    pk + (size_t)((((t * NKB + kb) * 16 + cbi) * 32 + lane) * 16);
                union { v16bf v; uint4 q[2]; } ua;
                ua.q[0] = *(const uint4*)(ap);
                ua.q[1] = *(const uint4*)(ap + 8);
                af[mi] = ua.v;
            }
            #pragma unroll
            for (int ni = 0; ni < NTILES; ni++) {
                const char* bp = lbase + bcol[ni] + tapoff;
                union { v16bf v; uint4 q[2]; } ub;
                ub.q[0] = *(const uint4*)(bp);
                ub.q[1] = *(const uint4*)(bp + 32);
                v16bf bv = ub.v;
                #pragma unroll
                for (int mi = 0; mi < 2; mi++) {
                    acc[mi][ni] = __builtin_amdgcn_wmma_f32_16x16x32_bf16(
                        false, af[mi], false, bv,
                        (short)0, acc[mi][ni], false, false);
                }
            }
        }
    };

    // ---- pipelined cin-block loop: loads for kb+1 hide under 126 WMMAs ----
    load_stage(0);
    store_stage(0);
    __syncthreads();
    for (int kb = 0; kb < NKB; kb++) {
        int buf = kb & 1;
        if (kb + 1 < NKB) load_stage(kb + 1);
        compute(kb, buf);
        if (kb + 1 < NKB) store_stage(1 - buf);
        __syncthreads();
    }

    // ---- epilogue: D lane%16 = spatial col, VGPR r = cout row (+8 if lane>=16) ----
    float* obase = out + (size_t)nimg * C_OUT * HW;
    #pragma unroll
    for (int ni = 0; ni < NTILES; ni++) {
        int c  = ni * 16 + (lane & 15);
        int yl = (c >= WW) ? 1 : 0;
        int xl = c - WW * yl;
        float* op = obase + (y0 + yl) * WW + xl;
        #pragma unroll
        for (int mi = 0; mi < 2; mi++) {
            int cobase = wave * 32 + mi * 16 + (lane >> 4) * 8;
            #pragma unroll
            for (int r = 0; r < 8; r++)
                op[(size_t)(cobase + r) * HW] = acc[mi][ni][r];
        }
    }
}

extern "C" void kernel_launch(void* const* d_in, const int* in_sizes, int n_in,
                              void* d_out, int out_size, void* d_ws, size_t ws_size,
                              hipStream_t stream) {
    const float* x = (const float*)d_in[0];      // (32,256,56,56) f32
    const float* W = (const float*)d_in[1];      // (256,256,3,3)  f32
    float* out = (float*)d_out;                  // (32,256,56,56) f32
    unsigned short* pk = (unsigned short*)d_ws;  // 1.18 MB packed bf16 weights

    binarize_pack_kernel<<<(KTAPS * NKB * 16 * 32 * 16) / 256, 256, 0, stream>>>(W, pk);

    dim3 grid(NIMG * (HH / TROWS));              // 896 blocks
    bconv_wmma_kernel<<<grid, dim3(256), 0, stream>>>(x, pk, out);
}
